// SegDiceLoss_79078937854483
// MI455X (gfx1250) — matrix operand order repaired and checked by reading, not compiled, not run on hardware
//
#include <hip/hip_runtime.h>

typedef __attribute__((ext_vector_type(16))) _Float16 v16h;
typedef __attribute__((ext_vector_type(2)))  __fp16   v2fp;
typedef __attribute__((ext_vector_type(8)))  float    v8f;

#define HW      16384
#define WIMG    128
#define CCH     16
#define PPAR    593
#define NCHUNK  4

__device__ __forceinline__ int pack2_f16(float a, float b) {
    union { v2fp v; int i; } u;
    u.v = __builtin_amdgcn_cvt_pkrtz(a, b);   // v_cvt_pk_rtz_f16_f32
    return u.i;
}

// ---------------------------------------------------------------------------
// Prepack: seg_feat [8,16,HW] f32 (channel-major) -> pk [8,HW,16] f16
// (pixel-major, 32B record = exactly one WMMA-B half-lane fragment, K=0..15).
// ---------------------------------------------------------------------------
__global__ __launch_bounds__(256) void seg_prepack(const float* __restrict__ seg,
                                                   v16h* __restrict__ pk)
{
    const int idx = blockIdx.x * 256 + threadIdx.x;   // b*HW + p
    const int b = idx >> 14;
    const int p = idx & (HW - 1);
    union { int d[8]; v16h v; } u;
    #pragma unroll
    for (int c = 0; c < 8; ++c)
        u.d[c] = pack2_f16(seg[(b * CCH + 2 * c) * HW + p],
                           seg[(b * CCH + 2 * c + 1) * HW + p]);
    pk[idx] = u.v;
}

// ---------------------------------------------------------------------------
// Main kernel: one block = (instance n, pixel chunk). 256 threads = 8 waves.
// K-order convention (A and B agree): K=0..15 -> feature channels 0..15,
// K=16 -> relx, K=17 -> rely, K=18..31 -> zero pad (in A; B is don't-care
// there since 0 x finite = 0).
// ---------------------------------------------------------------------------
__global__ __launch_bounds__(256) void seg_dice_main(
    const v16h* __restrict__ pk,          // [8, HW] packed f16 feature records
    const float* __restrict__ cw,         // [8, 593, HW]
    const long long* __restrict__ ind,    // [8, 32]
    const float* __restrict__ target,     // [256, HW]
    float* __restrict__ partials)         // [256, NCHUNK, 3]
{
    const int n    = blockIdx.x;
    const int chunk= blockIdx.y;
    const int b    = n >> 5;
    const int tid  = threadIdx.x;
    const int lane = tid & 31;
    const int wave = tid >> 5;
    const int m    = lane & 15;           // WMMA column / row-in-half
    const int hi   = lane >> 4;           // half-wave id

    __shared__ float sp[PPAR];
    __shared__ float red[8][3];

    const int indv = (int)ind[n];
    for (int p = tid; p < PPAR; p += 256)
        sp[p] = cw[(b * PPAR + p) * HW + indv];
    __syncthreads();

    // --- A fragments. A layout: lane l, half h ->
    //     (M = l&15, K = (l>>4)*16 + h)
    // w0 is [16 x 18], input i: 0,1 = coords, 2..17 = features.
    // K remap: K<16 -> i=K+2 ; K==16 -> i=0 ; K==17 -> i=1 ; else 0.
    v16h A0, A1;
    #pragma unroll
    for (int h = 0; h < 16; ++h) {
        const int K = (hi << 4) + h;
        float a0 = 0.f, a1 = 0.f;
        if (K < 16)      a0 = sp[m * 18 + K + 2];
        else if (K < 18) a0 = sp[m * 18 + (K - 16)];
        if (K < 16)      a1 = sp[288 + m * 16 + K];
        A0[h] = (_Float16)a0;
        A1[h] = (_Float16)a1;
    }
    v8f C0, C1;                           // biases pre-loaded as C operand
    float w2r[8];
    #pragma unroll
    for (int r = 0; r < 8; ++r) {
        C0[r]  = sp[560 + r + 8 * hi];
        C1[r]  = sp[576 + r + 8 * hi];
        w2r[r] = sp[544 + r + 8 * hi];
    }
    const float b2 = sp[592];
    const float fx = (float)(indv & (WIMG - 1));
    const float fy = (float)(indv >> 7);

    float ai = 0.f, as2 = 0.f, at2 = 0.f;

    const int t0 = chunk * 256;
    #pragma unroll 2
    for (int t = t0 + wave; t < t0 + 256; t += 8) {
        const int p = t * 16 + m;         // this lane's pixel (column N = m)

        // --- Layer-0 B operand: every lane loads the packed feature record
        //     (hi=1 redundant but L0-resident); hi=1 lanes replace dword 0
        //     with packed rel-coords (K=16,17). Remaining hi=1 dwords feed
        //     K=18..31 where A0 == 0 -> don't-care.
        union { int d[8]; v16h v; } ux;
        ux.v = pk[b * HW + p];
        const int cpack = pack2_f16(fx - (float)(p & (WIMG - 1)),
                                    fy - (float)(p >> 7));
        ux.d[0] = hi ? cpack : ux.d[0];

        v8f d0 = __builtin_amdgcn_wmma_f32_16x16x32_f16(
            false, A0, false, ux.v, (short)0, C0, false, false);

        // ReLU -> 4 packed f16 dwords; shuffle packed dwords across the
        // half-waves; assemble layer-1 B operand. hi=1 lanes carry finite
        // garbage at K=16..31 where A1 == 0 -> don't-care, no masking.
        int own[4], par[4];
        #pragma unroll
        for (int r = 0; r < 4; ++r)
            own[r] = pack2_f16(fmaxf(d0[2 * r],     0.f),
                               fmaxf(d0[2 * r + 1], 0.f));
        #pragma unroll
        for (int r = 0; r < 4; ++r) par[r] = __shfl_xor(own[r], 16, 32);
        union { int d[8]; v16h v; } ub;
        #pragma unroll
        for (int r = 0; r < 4; ++r) {
            ub.d[r]     = own[r];          // K = 0..7   (own rows)
            ub.d[4 + r] = par[r];          // K = 8..15  (partner rows)
        }

        v8f d1 = __builtin_amdgcn_wmma_f32_16x16x32_f16(
            false, A1, false, ub.v, (short)0, C1, false, false);

        // Layer 2: 1x16 dot split across the two half-waves.
        float partial = 0.f;
        #pragma unroll
        for (int r = 0; r < 8; ++r)
            partial = fmaf(w2r[r], fmaxf(d1[r], 0.f), partial);
        const float logit = partial + __shfl_xor(partial, 16, 32) + b2;
        const float s  = __builtin_amdgcn_rcpf(1.f + __expf(-logit));
        const float tg = target[n * HW + p];
        ai  = fmaf(s, tg, ai);
        as2 = fmaf(s, s, as2);
        at2 = fmaf(tg, tg, at2);
    }

    // Wave reduction (half-waves hold duplicates -> corrected by 0.5 below).
    #pragma unroll
    for (int off = 16; off >= 1; off >>= 1) {
        ai  += __shfl_xor(ai,  off, 32);
        as2 += __shfl_xor(as2, off, 32);
        at2 += __shfl_xor(at2, off, 32);
    }
    if (lane == 0) { red[wave][0] = ai; red[wave][1] = as2; red[wave][2] = at2; }
    __syncthreads();
    if (tid == 0) {
        float si = 0.f, ss = 0.f, st = 0.f;
        for (int w = 0; w < 8; ++w) { si += red[w][0]; ss += red[w][1]; st += red[w][2]; }
        float* out = &partials[(n * NCHUNK + chunk) * 3];
        out[0] = 0.5f * si;  out[1] = 0.5f * ss;  out[2] = 0.5f * st;
    }
}

// ---------------------------------------------------------------------------
// Finalize: per-instance dice loss, mask, sum / B (fixed order, no atomics).
// ---------------------------------------------------------------------------
__global__ __launch_bounds__(256) void seg_dice_finalize(
    const float* __restrict__ partials,
    const unsigned char* __restrict__ mask,
    float* __restrict__ out)
{
    __shared__ float buf[8];
    const int n = threadIdx.x;            // instance 0..255
    float inter = 0.f, s2 = 0.f, t2 = 0.f;
    for (int c = 0; c < NCHUNK; ++c) {
        const float* pp = &partials[(n * NCHUNK + c) * 3];
        inter += pp[0]; s2 += pp[1]; t2 += pp[2];
    }
    const float loss = 1.f - 2.f * inter / (s2 + t2 + 1e-5f);
    float v = mask[n] ? loss : 0.f;
    #pragma unroll
    for (int off = 1; off < 32; off <<= 1) v += __shfl_xor(v, off, 32);
    if ((n & 31) == 0) buf[n >> 5] = v;
    __syncthreads();
    if (n == 0) {
        float s = 0.f;
        for (int w = 0; w < 8; ++w) s += buf[w];
        out[0] = s * 0.125f;              // / B
    }
}

extern "C" void kernel_launch(void* const* d_in, const int* in_sizes, int n_in,
                              void* d_out, int out_size, void* d_ws, size_t ws_size,
                              hipStream_t stream) {
    (void)in_sizes; (void)n_in; (void)out_size; (void)ws_size;
    const float*         seg    = (const float*)d_in[0];
    const float*         cw     = (const float*)d_in[1];
    const unsigned char* mask   = (const unsigned char*)d_in[2];
    const long long*     ind    = (const long long*)d_in[3];
    const float*         target = (const float*)d_in[4];

    float* partials = (float*)d_ws;                        // 12 KB, all written
    v16h*  pkbuf    = (v16h*)((char*)d_ws + 16384);        // 8*HW*32B = 4 MB

    seg_prepack<<<(8 * HW) / 256, 256, 0, stream>>>(seg, pkbuf);
    seg_dice_main<<<dim3(256, NCHUNK), 256, 0, stream>>>(pkbuf, cw, ind, target, partials);
    seg_dice_finalize<<<1, 256, 0, stream>>>(partials, mask, (float*)d_out);
}
